// SelfAttention_27075473834365
// MI455X (gfx1250) — matrix operand order repaired and checked by reading, not compiled, not run on hardware
//
#include <hip/hip_runtime.h>

#define AS1 __attribute__((address_space(1)))
#define AS3 __attribute__((address_space(3)))

#if defined(__has_builtin)
#if __has_builtin(__builtin_amdgcn_global_store_async_from_lds_b128) && \
    __has_builtin(__builtin_amdgcn_s_wait_asynccnt)
#define USE_ASYNC_D_STORE 1
#endif
#endif

typedef __attribute__((ext_vector_type(16))) _Float16 v16h;
typedef __attribute__((ext_vector_type(4)))  _Float16 v4h;
typedef __attribute__((ext_vector_type(8)))  float    v8f;
typedef __attribute__((ext_vector_type(4)))  int      v4i;

constexpr int BB = 16;    // batch
constexpr int SS = 2048;  // sequence length
constexpr int HH = 1024;  // hidden
constexpr int AA = 1024;  // attention dim

static __device__ __forceinline__ v8f wmma_f16(v16h a, v16h b, v8f c) {
    return __builtin_amdgcn_wmma_f32_16x16x32_f16(false, a, false, b, (short)0, c,
                                                  false, false);
}

// ---------------- fp32 -> f16 cast, 4-wide (b128 in, b64 out) ----------------
__global__ void cast_kernel(const float4* __restrict__ src, v4h* __restrict__ dst,
                            int n4) {
    for (int i = blockIdx.x * blockDim.x + threadIdx.x; i < n4;
         i += gridDim.x * blockDim.x) {
        float4 f = src[i];
        v4h h;
        h[0] = (_Float16)f.x; h[1] = (_Float16)f.y;
        h[2] = (_Float16)f.z; h[3] = (_Float16)f.w;
        dst[i] = h;
    }
}

// ---------------- projection GEMM: C = act(A[M,K] @ W[K,N]) as f16 ----------
// M = BB*SS = 32768, K = HH, N = AA.  256 threads = 8 waves.
__global__ __launch_bounds__(256, 1)
void proj_gemm(const _Float16* __restrict__ Ah, const _Float16* __restrict__ Wh,
               _Float16* __restrict__ Ch, int do_tanh) {
    const int wave = threadIdx.x >> 5;
    const int lane = threadIdx.x & 31;
    const int m0 = blockIdx.x * 64 + (wave & 3) * 16;
    const int n0 = blockIdx.y * 128 + (wave >> 2) * 64;
    const int K = HH, N = AA;

    const int ar = lane & 15;
    const int ac = (lane >> 4) * 16;
    v8f acc[4] = {};

    for (int k0 = 0; k0 < K; k0 += 32) {
        v16h af = *(const v16h*)(Ah + (size_t)(m0 + ar) * K + k0 + ac);
#pragma unroll
        for (int nt = 0; nt < 4; ++nt) {
            v16h bf;
#pragma unroll
            for (int s = 0; s < 16; ++s)
                bf[s] = Wh[(size_t)(k0 + ac + s) * N + n0 + nt * 16 + ar];
            acc[nt] = wmma_f16(af, bf, acc[nt]);
        }
    }
    const int nn = lane & 15;
    const int mb = (lane >> 4) * 8;
#pragma unroll
    for (int nt = 0; nt < 4; ++nt)
#pragma unroll
        for (int i = 0; i < 8; ++i) {
            float v = acc[nt][i];
            if (do_tanh) v = tanhf(v);
            Ch[(size_t)(m0 + mb + i) * N + n0 + nt * 16 + nn] = (_Float16)v;
        }
}

// ---------------- fused causal attention per 16-row query block -------------
// grid = (S/16, B), block = 256 threads (8 waves).
__global__ __launch_bounds__(256, 1)
void attn_kernel(const _Float16* __restrict__ xh, const _Float16* __restrict__ Kh,
                 const _Float16* __restrict__ Qh, float* __restrict__ a_out,
                 float* __restrict__ d_out) {
    __shared__ float sc[16][SS];    // 128 KB score/prob buffer
    __shared__ float red[16][16];   // softmax partials

    const int tid  = threadIdx.x;
    const int wave = tid >> 5;
    const int lane = tid & 31;
    const int qblk = blockIdx.x;    // 0..127
    const int b    = blockIdx.y;    // 0..15
    const int row0 = qblk * 16;

    const _Float16* Qb = Qh + ((size_t)b * SS + row0) * AA;
    const _Float16* Kb = Kh + (size_t)b * SS * AA;
    const _Float16* Xb = xh + (size_t)b * SS * HH;

    const int ar = lane & 15;
    const int ac = (lane >> 4) * 16;
    const int nn = lane & 15;
    const int mb = (lane >> 4) * 8;

    // ---- phase 1: scores = Q_blk @ K^T. 4 column tiles share one A-fragment.
    // Full groups take the branch-free pipelined path; only the diagonal group
    // of each wave falls into the remainder path.
    for (int g = wave * 4; g <= qblk; g += 32) {
        v8f c[4] = {};
        if (qblk - g >= 3) {                       // full group of 4 tiles
            for (int k0 = 0; k0 < AA; k0 += 32) {
                v16h af = *(const v16h*)(Qb + (size_t)ar * AA + k0 + ac);
#pragma unroll
                for (int j = 0; j < 4; ++j) {
                    v16h bf = *(const v16h*)(Kb + (size_t)((g + j) * 16 + ar) * AA +
                                             k0 + ac);
                    c[j] = wmma_f16(af, bf, c[j]);
                }
            }
#pragma unroll
            for (int j = 0; j < 4; ++j)
#pragma unroll
                for (int i = 0; i < 8; ++i)
                    sc[mb + i][(g + j) * 16 + nn] = c[j][i];
        } else {                                   // partial (diagonal) group
            const int nvalid = qblk - g + 1;       // 1..3, wave-uniform
            for (int k0 = 0; k0 < AA; k0 += 32) {
                v16h af = *(const v16h*)(Qb + (size_t)ar * AA + k0 + ac);
                for (int j = 0; j < nvalid; ++j) {
                    v16h bf = *(const v16h*)(Kb + (size_t)((g + j) * 16 + ar) * AA +
                                             k0 + ac);
                    c[j] = wmma_f16(af, bf, c[j]);
                }
            }
            for (int j = 0; j < nvalid; ++j)
#pragma unroll
                for (int i = 0; i < 8; ++i)
                    sc[mb + i][(g + j) * 16 + nn] = c[j][i];
        }
    }
    __syncthreads();

    // ---- phase 2: exact softmax over valid prefix, zero masked tail ----
    {
        const int r   = tid >> 4;        // 16 rows, 16 threads per row
        const int seg = tid & 15;
        const int sg  = row0 + r;        // causal bound
        const int tb  = seg * (SS / 16);
        const int te  = tb + (SS / 16);
        const int lim = (sg + 1 < te) ? ((sg + 1 > tb) ? sg + 1 : tb) : te;

        float mx = -__builtin_inff();
        for (int t = tb; t < lim; ++t) mx = fmaxf(mx, sc[r][t]);
        red[r][seg] = mx;
        __syncthreads();
        float rm = -__builtin_inff();
#pragma unroll
        for (int j = 0; j < 16; ++j) rm = fmaxf(rm, red[r][j]);
        __syncthreads();

        float sum = 0.f;
        for (int t = tb; t < lim; ++t) {
            float e = expf(sc[r][t] - rm);
            sc[r][t] = e;
            sum += e;
        }
        red[r][seg] = sum;
        __syncthreads();
        float rs = 0.f;
#pragma unroll
        for (int j = 0; j < 16; ++j) rs += red[r][j];
        const float inv = 1.0f / rs;
        for (int t = tb; t < lim; ++t) sc[r][t] *= inv;
        for (int t = lim; t < te; ++t) sc[r][t] = 0.0f;   // masked tail
        __syncthreads();
    }

    // ---- write d: stream the LDS buffer to HBM ----
    {
        float* drow = d_out + ((size_t)b * SS + row0) * SS;
#if defined(USE_ASYNC_D_STORE)
        // CDNA5 async LDS->global path (ASYNCcnt-tracked, no VGPR staging).
        for (int i = tid * 4; i < 16 * SS; i += 256 * 4) {
            const int r = i >> 11;
            const int t = i & (SS - 1);
            __builtin_amdgcn_global_store_async_from_lds_b128(
                (AS1 v4i*)(drow + (size_t)r * SS + t), (AS3 v4i*)&sc[r][t], 0, 0);
        }
#else
        for (int i = tid; i < 16 * SS; i += 256) {
            const int r = i >> 11;
            const int t = i & (SS - 1);
            drow[(size_t)r * SS + t] = sc[r][t];
        }
#endif
    }

    // ---- phase 3: a_blk = P @ x_b, wave owns 128 output cols (8 tiles) ----
    {
        const int nw = wave * 128;
        const int kend = (row0 + 16 + 31) & ~31;  // causal prefix, 32-aligned
        v8f acc[8] = {};
        for (int k0 = 0; k0 < kend; k0 += 32) {
            v16h af;
#pragma unroll
            for (int j = 0; j < 16; ++j)
                af[j] = (_Float16)sc[ar][k0 + ac + j];
#pragma unroll
            for (int nt = 0; nt < 8; ++nt) {
                v16h bf;
#pragma unroll
                for (int s = 0; s < 16; ++s)
                    bf[s] = Xb[(size_t)(k0 + ac + s) * HH + nw + nt * 16 + ar];
                acc[nt] = wmma_f16(af, bf, acc[nt]);
            }
        }
#pragma unroll
        for (int nt = 0; nt < 8; ++nt)
#pragma unroll
            for (int i = 0; i < 8; ++i)
                a_out[((size_t)b * SS + row0 + mb + i) * HH + nw + nt * 16 + nn] =
                    acc[nt][i];
    }

#if defined(USE_ASYNC_D_STORE)
    __builtin_amdgcn_s_wait_asynccnt(0);  // drain async d stores before exit
#endif
}

extern "C" void kernel_launch(void* const* d_in, const int* in_sizes, int n_in,
                              void* d_out, int out_size, void* d_ws, size_t ws_size,
                              hipStream_t stream) {
    (void)in_sizes; (void)n_in; (void)out_size; (void)ws_size;

    const float* x  = (const float*)d_in[0];   // [B,S,H]
    const float* wa = (const float*)d_in[1];   // [H,A]
    const float* qw = (const float*)d_in[2];   // [H,A]

    float* a_out = (float*)d_out;                     // [B,S,H]
    float* d_mat = a_out + (size_t)BB * SS * HH;      // [B,S,S]

    // workspace carve-up (f16): x, w_a, query, K=tanh(xW), Q=xWq  (~206 MB)
    _Float16* xh  = (_Float16*)d_ws;
    _Float16* wah = xh  + (size_t)BB * SS * HH;
    _Float16* qwh = wah + (size_t)HH * AA;
    _Float16* Kh  = qwh + (size_t)HH * AA;
    _Float16* Qh  = Kh  + (size_t)BB * SS * AA;

    cast_kernel<<<4096, 256, 0, stream>>>((const float4*)x,  (v4h*)xh,
                                          BB * SS * HH / 4);
    cast_kernel<<<512, 256, 0, stream>>>((const float4*)wa, (v4h*)wah,
                                         HH * AA / 4);
    cast_kernel<<<512, 256, 0, stream>>>((const float4*)qw, (v4h*)qwh,
                                         HH * AA / 4);

    dim3 pg((BB * SS) / 64, AA / 128);
    proj_gemm<<<pg, 256, 0, stream>>>(xh, wah, Kh, 1);  // K = tanh(x @ w_a)
    proj_gemm<<<pg, 256, 0, stream>>>(xh, qwh, Qh, 0);  // Q = x @ query

    attn_kernel<<<dim3(SS / 16, BB), 256, 0, stream>>>(xh, Kh, Qh, a_out, d_mat);
}
